// RelationalSubgraphEncoder2_48344151884371
// MI455X (gfx1250) — matrix operand order repaired and checked by reading, not compiled, not run on hardware
//
#include <hip/hip_runtime.h>
#include <hip/hip_bf16.h>

#define N_SRC 50000
#define N_DST 50000
#define NEDGE 800000
#define IN_F  256
#define NH    8
#define ND    16
#define OUTF  (NH * ND)     // 128
#define NEG_SLOPE 0.2f

typedef __attribute__((ext_vector_type(2)))  float    v2f;
typedef __attribute__((ext_vector_type(8)))  float    v8f;
typedef __attribute__((ext_vector_type(16))) _Float16 v16h;

#if __has_builtin(__builtin_amdgcn_wmma_f32_16x16x4_f32)
#define USE_F32_WMMA 1
#else
#define USE_F32_WMMA 0
#endif

// ---------------------------------------------------------------------------
// order-preserving float <-> uint map for atomic segment-max
// ---------------------------------------------------------------------------
__device__ __forceinline__ unsigned f2ord(float x) {
    unsigned b = __float_as_uint(x);
    return (b & 0x80000000u) ? ~b : (b | 0x80000000u);
}
__device__ __forceinline__ float ord2f(unsigned u) {
    unsigned b = (u & 0x80000000u) ? (u & 0x7FFFFFFFu) : ~u;
    return __uint_as_float(b);
}

// ---------------------------------------------------------------------------
// rw = rel_emb @ W_rel   (64 x [64,256] -> 256)
// ---------------------------------------------------------------------------
__global__ __launch_bounds__(256) void rw_kernel(const float* __restrict__ rel,
                                                 const float* __restrict__ Wrel,
                                                 float* __restrict__ rw) {
    int c = threadIdx.x;
    float s = 0.f;
#pragma unroll 8
    for (int i = 0; i < 64; ++i) s += rel[i] * Wrel[i * 256 + c];
    rw[c] = s;
}

// ---------------------------------------------------------------------------
// Fused projection GEMM + attention-logit epilogue.
//   feat [N,256] @ W [256,128] -> (optional) feat_out [N,128]
//   logit[n,h] = sum_d out[n, h*16+d] * rw[h*32 + att_off + d]
// Block = 256 threads = 8 waves; block owns 16 rows; wave w owns cols
// [16w,16w+16) == head w.  WMMA f32 16x16x4 (exact) or f16 16x16x32 fallback.
// ---------------------------------------------------------------------------
__global__ __launch_bounds__(256) void gemm_logits_kernel(
    const float* __restrict__ feat, const float* __restrict__ W,
    const float* __restrict__ rw, int att_off,
    float* __restrict__ feat_out, float* __restrict__ logit_out) {

    __shared__ float As[16][260];   // 260-pad: lane m hits bank (4m + k) % 64

    const int tid  = threadIdx.x;
    const int base = blockIdx.x * 16;

    // stage A strip: 16 rows x 256 f32 via float4, fully coalesced
    const float4* src4 = (const float4*)(feat + (size_t)base * IN_F);
#pragma unroll
    for (int i = 0; i < 4; ++i) {
        int idx = tid + i * 256;        // 0..1023 float4 slots
        int row = idx >> 6;
        int c4  = idx & 63;
        float4 v = src4[row * 64 + c4];
        As[row][c4 * 4 + 0] = v.x;
        As[row][c4 * 4 + 1] = v.y;
        As[row][c4 * 4 + 2] = v.z;
        As[row][c4 * 4 + 3] = v.w;
    }
    __syncthreads();

    const int wave = tid >> 5;
    const int lane = tid & 31;
    const int n    = lane & 15;     // output column within tile / M-row for A
    const int half = lane >> 4;
    const int cb   = wave * 16;     // column base == head * D
    const int col  = cb + n;

    v8f acc = {};

#if USE_F32_WMMA
    // A 16x4 f32: lane(0-15) M=lane, VGPR0/1 = K0/K1 ; lane(16-31) K2/K3
    // B 4x16 f32: mirrored with N instead of M
    for (int k0 = 0; k0 < IN_F; k0 += 4) {
        const int ka = k0 + half * 2;
        v2f a, b;
        a[0] = As[n][ka];
        a[1] = As[n][ka + 1];
        const float* Bp = W + (size_t)ka * OUTF + col;
        b[0] = Bp[0];
        b[1] = Bp[OUTF];
        acc = __builtin_amdgcn_wmma_f32_16x16x4_f32(
            false, a, false, b, (short)0, acc, false, false);
    }
#else
    // f16 fallback: A 16x32 / B 32x16, both in contiguous k = 16*half + s order
    for (int k0 = 0; k0 < IN_F; k0 += 32) {
        v16h a, b;
#pragma unroll
        for (int j = 0; j < 8; ++j) {
            int kaj = k0 + 16 * half + 2 * j;
            a[2 * j]     = (_Float16)As[n][kaj];
            a[2 * j + 1] = (_Float16)As[n][kaj + 1];
            int kbj = k0 + 16 * half + 2 * j;
            b[2 * j]     = (_Float16)W[(size_t)kbj * OUTF + col];
            b[2 * j + 1] = (_Float16)W[(size_t)(kbj + 1) * OUTF + col];
        }
        acc = __builtin_amdgcn_wmma_f32_16x16x32_f16(
            false, a, false, b, (short)0, acc, false, false);
    }
#endif

    // C/D layout: VGPR r -> M = r + 8*half, N = n
    if (feat_out) {
#pragma unroll
        for (int r = 0; r < 8; ++r) {
            int m = r + half * 8;
            feat_out[(size_t)(base + m) * OUTF + col] = acc[r];
        }
    }

    // attention logit: dot over d (= lane group of 16) for each of 8 rows
    const float attv = rw[wave * 32 + att_off + n];
    float p[8];
#pragma unroll
    for (int r = 0; r < 8; ++r) p[r] = acc[r] * attv;
#pragma unroll
    for (int mask = 1; mask < 16; mask <<= 1) {
#pragma unroll
        for (int r = 0; r < 8; ++r) p[r] += __shfl_xor(p[r], mask, 32);
    }
    if (n == 0) {
#pragma unroll
        for (int r = 0; r < 8; ++r) {
            int m = r + half * 8;
            logit_out[(size_t)(base + m) * NH + wave] = p[r];
        }
    }
}

// ---------------------------------------------------------------------------
// init: zero d_out & dsum, set dmax to ord(-inf)
// ---------------------------------------------------------------------------
__global__ __launch_bounds__(256) void init_kernel(float* __restrict__ out,
                                                   unsigned* __restrict__ dmax,
                                                   float* __restrict__ dsum) {
    int t = blockIdx.x * blockDim.x + threadIdx.x;
    if (t < N_DST * OUTF) out[t] = 0.f;
    if (t < N_DST * NH) {
        dmax[t] = 0x007FFFFFu;   // f2ord(-inf)
        dsum[t] = 0.f;
    }
}

// ---------------------------------------------------------------------------
// edge pass 1: e = leaky_relu(e_src[s,h] + e_dst[d,h]); segment max over dst
// ---------------------------------------------------------------------------
__global__ __launch_bounds__(256) void edge1_kernel(
    const int* __restrict__ esrc, const int* __restrict__ edst,
    const float* __restrict__ lsrc, const float* __restrict__ ldst,
    float* __restrict__ ew, unsigned* __restrict__ dmax) {
    int t = blockIdx.x * blockDim.x + threadIdx.x;
    if (t >= NEDGE * NH) return;
    int e = t >> 3, h = t & 7;
    int s = esrc[e], d = edst[e];
    float v = lsrc[s * NH + h] + ldst[d * NH + h];
    v = (v > 0.f) ? v : NEG_SLOPE * v;
    ew[t] = v;
    atomicMax(&dmax[d * NH + h], f2ord(v));
}

// ---------------------------------------------------------------------------
// edge pass 2: a = exp(e - max); segment sum over dst
// ---------------------------------------------------------------------------
__global__ __launch_bounds__(256) void edge2_kernel(
    const int* __restrict__ edst, float* __restrict__ ew,
    const unsigned* __restrict__ dmax, float* __restrict__ dsum) {
    int t = blockIdx.x * blockDim.x + threadIdx.x;
    if (t >= NEDGE * NH) return;
    int e = t >> 3, h = t & 7;
    int d = edst[e];
    float a = expf(ew[t] - ord2f(dmax[d * NH + h]));
    ew[t] = a;
    atomicAdd(&dsum[d * NH + h], a);
}

// ---------------------------------------------------------------------------
// edge pass 3: out[dst, h*16+dd] += fs[src, h*16+dd] * a / dsum[dst,h]
// thread per (edge, channel): coalesced fs gather + coalesced atomics
// ---------------------------------------------------------------------------
__global__ __launch_bounds__(256) void edge3_kernel(
    const int* __restrict__ esrc, const int* __restrict__ edst,
    const float* __restrict__ fs, const float* __restrict__ ew,
    const float* __restrict__ dsum, float* __restrict__ out) {
    int t = blockIdx.x * blockDim.x + threadIdx.x;
    if (t >= NEDGE * OUTF) return;
    int e = t >> 7, c = t & 127;
    int h = c >> 4;
    int s = esrc[e], d = edst[e];
    float coef = ew[e * NH + h] / dsum[d * NH + h];
    atomicAdd(&out[(size_t)d * OUTF + c], fs[(size_t)s * OUTF + c] * coef);
}

// ---------------------------------------------------------------------------
// final relu in place on d_out
// ---------------------------------------------------------------------------
__global__ __launch_bounds__(256) void relu_kernel(float* __restrict__ out) {
    int t = blockIdx.x * blockDim.x + threadIdx.x;
    if (t < N_DST * OUTF) out[t] = fmaxf(out[t], 0.f);
}

// ---------------------------------------------------------------------------
extern "C" void kernel_launch(void* const* d_in, const int* in_sizes, int n_in,
                              void* d_out, int out_size, void* d_ws, size_t ws_size,
                              hipStream_t stream) {
    const float* feat_src = (const float*)d_in[0];
    const float* feat_dst = (const float*)d_in[1];
    const float* W_src    = (const float*)d_in[2];
    const float* W_dst    = (const float*)d_in[3];
    const float* rel_emb  = (const float*)d_in[4];
    const float* W_rel    = (const float*)d_in[5];
    const int*   edge_src = (const int*)d_in[6];
    const int*   edge_dst = (const int*)d_in[7];
    float* out = (float*)d_out;

    float* ws    = (float*)d_ws;
    float* rw    = ws;                       // 256
    float* fs    = rw + 256;                 // N_SRC*128
    float* lsrc  = fs + (size_t)N_SRC * OUTF;   // N_SRC*8
    float* ldst  = lsrc + (size_t)N_SRC * NH;   // N_DST*8
    float* ew    = ldst + (size_t)N_DST * NH;   // E*8
    unsigned* dmax = (unsigned*)(ew + (size_t)NEDGE * NH); // N_DST*8
    float* dsum  = (float*)(dmax + (size_t)N_DST * NH);    // N_DST*8

    rw_kernel<<<1, 256, 0, stream>>>(rel_emb, W_rel, rw);

    // src projection: store fs + e_src (rw offset 16 = second half of [H,2D])
    gemm_logits_kernel<<<N_SRC / 16, 256, 0, stream>>>(
        feat_src, W_src, rw, ND, fs, lsrc);
    // dst projection: only logits needed (rw offset 0)
    gemm_logits_kernel<<<N_DST / 16, 256, 0, stream>>>(
        feat_dst, W_dst, rw, 0, nullptr, ldst);

    init_kernel<<<(N_DST * OUTF + 255) / 256, 256, 0, stream>>>(out, dmax, dsum);

    edge1_kernel<<<(NEDGE * NH + 255) / 256, 256, 0, stream>>>(
        edge_src, edge_dst, lsrc, ldst, ew, dmax);
    edge2_kernel<<<(NEDGE * NH + 255) / 256, 256, 0, stream>>>(
        edge_dst, ew, dmax, dsum);
    edge3_kernel<<<(NEDGE * OUTF + 255) / 256, 256, 0, stream>>>(
        edge_src, edge_dst, fs, ew, dsum, out);

    relu_kernel<<<(N_DST * OUTF + 255) / 256, 256, 0, stream>>>(out);
}
